// PatchedWSA_24567212933746
// MI455X (gfx1250) — matrix operand order repaired and checked by reading, not compiled, not run on hardware
//
#include <hip/hip_runtime.h>
#include <hip/hip_bf16.h>
#include <math.h>

// ---------------------------------------------------------------------------
// PatchedWSA forward for MI455X (gfx1250).  Heavy GEMMs use
// v_wmma_f32_16x16x32_f16 with 128x64x32 block tiles, double-buffered LDS,
// branch-free staging (vectorized fast path + clamped/select edge path).
// ---------------------------------------------------------------------------

typedef __attribute__((ext_vector_type(16))) _Float16 v16h;
typedef __attribute__((ext_vector_type(4)))  _Float16 v4h;
typedef __attribute__((ext_vector_type(8)))  float    v8f;

#define BSZ   8
#define SEQL  8192
#define PREDN 96
#define DM    512
#define HH_   8
#define NP    1023
#define LW    511          // DWT length
#define DFF_  2048

static constexpr long long LBD    = (long long)LW * DM;      // 261632 per-batch rows*D
static constexpr long long MROWS  = (long long)BSZ * LW;     // 4088
static constexpr long long PT_SZ  = (long long)DM * NP;      // 523776 per batch

// d_out offsets (floats): preds | patch_recon | patches | LL | LH
static constexpr long long O_PRED  = 0;
static constexpr long long O_RECON = 768;
static constexpr long long O_PATCH = O_RECON + (long long)BSZ * DM * 1022;
static constexpr long long O_LL    = O_PATCH + (long long)BSZ * DM * NP;
static constexpr long long O_LH    = O_LL    + (long long)BSZ * DM * LW;

// ws offsets (floats) -- ~143 MB total with reuse
static constexpr long long W_XN    = 0;
static constexpr long long W_STATS = 65536;
static constexpr long long W_GATE  = 65568;
static constexpr long long W_LLSEQ = 70144;
static constexpr long long W_LHSEQ = W_LLSEQ + 8 * LBD;
static constexpr long long W_Q     = W_LHSEQ + 8 * LBD;
static constexpr long long W_K     = W_Q     + 8 * LBD;
static constexpr long long W_V     = W_K     + 8 * LBD;
static constexpr long long W_O     = W_V     + 8 * LBD;
static constexpr long long W_LLOUT = W_O     + 8 * LBD;
static constexpr long long W_LHOUT = W_LLOUT + 8 * LBD;
static constexpr long long W_CROSS = W_LHOUT + 8 * LBD;
static constexpr long long W_S     = W_CROSS + 8 * LBD;       // 64*511*512
// reuse aliases
static constexpr long long W_F1  = W_Q;     // LL_fused
static constexpr long long W_F2  = W_K;     // LH_fused
static constexpr long long W_HID = W_S;     // MLP hidden
static constexpr long long W_LLO = W_V;     // LL_o seq
static constexpr long long W_LHO = W_O;     // LH_o seq

__device__ __forceinline__ float gelu_f(float x) {
    return 0.5f * x * (1.0f + erff(x * 0.70710678118654752f));
}
__device__ __forceinline__ float sigmoid_f(float x) {
    return 1.0f / (1.0f + expf(-x));
}

// ---------------------------------------------------------------------------
// WMMA GEMM:  C = alpha * A(MxK) x B(KxN) [+bias][+resid][gelu]
// Block tile 128x64, K-step 32, 256 threads = 8 waves in a 4x2 grid; each
// wave computes a 32x32 C tile (2 A-frags x 2 B-frags -> 4 wmma / step).
// LDS double-buffered; staging is branch-free.
// Batched over blockIdx.z with split outer/inner strides (z = zo*HHd + zi).
// ---------------------------------------------------------------------------
#define BM 128
#define BN 64
#define BK 32
#define LSTR 36   // halves per LDS row (72 B: 8B-aligned rows, bank-spread)

__global__ __launch_bounds__(256) void gemm_wmma_kernel(
    const float* __restrict__ A, long long lda, long long sAo, long long sAi,
    const float* __restrict__ B, long long ldb, long long sBo, long long sBi, int transB,
    const float* __restrict__ bias,
    const float* __restrict__ resid,
    float* __restrict__ C, long long ldc, long long sCo, long long sCi,
    int M, int N, int K, float alpha, int act, int HHd)
{
    __shared__ _Float16 As[2][BM * LSTR];
    __shared__ _Float16 Bs[2][BN * LSTR];

    const int tid  = threadIdx.x;
    const int z    = blockIdx.z;
    const int zo   = z / HHd, zi = z % HHd;
    A += zo * sAo + zi * sAi;
    B += zo * sBo + zi * sBi;
    C += zo * sCo + zi * sCi;
    if (resid) resid += zo * sCo + zi * sCi;

    const int m0 = blockIdx.y * BM;
    const int n0 = blockIdx.x * BN;
    const int wave = tid >> 5;
    const int lane = tid & 31;
    const int wm = wave >> 1;            // 0..3 -> 32 rows each
    const int wn = wave & 1;             // 0..1 -> 32 cols each
    const int lr = lane & 15;
    const int lh = lane >> 4;

    // ---- staging (branch-free) ----
    auto stageA = [&](int buf, int k0) {
        if (m0 + BM <= M && k0 + BK <= K) {
            // fast: 4096 floats = 1024 float4, 4 per thread
            #pragma unroll
            for (int i = 0; i < 4; ++i) {
                int id4 = tid + i * 256;
                int r = id4 >> 3, c4 = (id4 & 7) << 2;
                const float4 v = *(const float4*)(A + (long long)(m0 + r) * lda + k0 + c4);
                v4h h; h.x = (_Float16)v.x; h.y = (_Float16)v.y;
                h.z = (_Float16)v.z; h.w = (_Float16)v.w;
                *(v4h*)(&As[buf][r * LSTR + c4]) = h;
                if (k0 + 2 * BK <= K)   // uniform-guarded prefetch of K+2 tile
                    __builtin_prefetch(A + (long long)(m0 + r) * lda + k0 + BK + c4, 0, 3);
            }
        } else {
            // edge: clamped address + value select, no control flow
            #pragma unroll
            for (int i = 0; i < 16; ++i) {
                int id = tid + i * 256;
                int r = id >> 5, c = id & 31;
                int gm = m0 + r, gk = k0 + c;
                int cm = gm < M ? gm : M - 1;
                int ck = gk < K ? gk : K - 1;
                float v = A[(long long)cm * lda + ck];
                As[buf][r * LSTR + c] = (_Float16)((gm < M && gk < K) ? v : 0.0f);
            }
        }
    };
    auto stageB = [&](int buf, int k0) {
        if (n0 + BN <= N && k0 + BK <= K) {
            if (transB) {
                // 64 rows(n) x 32 cols(k): 512 float4, 2 per thread
                #pragma unroll
                for (int i = 0; i < 2; ++i) {
                    int id4 = tid + i * 256;
                    int nn = id4 >> 3, k4 = (id4 & 7) << 2;
                    const float4 v = *(const float4*)(B + (long long)(n0 + nn) * ldb + k0 + k4);
                    v4h h; h.x = (_Float16)v.x; h.y = (_Float16)v.y;
                    h.z = (_Float16)v.z; h.w = (_Float16)v.w;
                    *(v4h*)(&Bs[buf][nn * LSTR + k4]) = h;
                }
            } else {
                // 32 rows(k) x 64 cols(n): vector load along n, scatter 4 b16
                #pragma unroll
                for (int i = 0; i < 2; ++i) {
                    int id4 = tid + i * 256;
                    int kk = id4 >> 4, n4 = (id4 & 15) << 2;
                    const float4 v = *(const float4*)(B + (long long)(k0 + kk) * ldb + n0 + n4);
                    Bs[buf][(n4 + 0) * LSTR + kk] = (_Float16)v.x;
                    Bs[buf][(n4 + 1) * LSTR + kk] = (_Float16)v.y;
                    Bs[buf][(n4 + 2) * LSTR + kk] = (_Float16)v.z;
                    Bs[buf][(n4 + 3) * LSTR + kk] = (_Float16)v.w;
                }
            }
        } else {
            #pragma unroll
            for (int i = 0; i < 8; ++i) {
                int id = tid + i * 256;
                int nn, kk;
                if (transB) { nn = id >> 5; kk = id & 31; }
                else        { kk = id >> 6; nn = id & 63; }
                int gn = n0 + nn, gk = k0 + kk;
                int cn = gn < N ? gn : N - 1;
                int ck = gk < K ? gk : K - 1;
                float v = transB ? B[(long long)cn * ldb + ck]
                                 : B[(long long)ck * ldb + cn];
                Bs[buf][nn * LSTR + kk] = (_Float16)((gn < N && gk < K) ? v : 0.0f);
            }
        }
    };

    v8f acc00 = {}, acc01 = {}, acc10 = {}, acc11 = {};

    stageA(0, 0); stageB(0, 0);
    int p = 0;
    for (int k0 = 0; k0 < K; k0 += BK) {
        __syncthreads();                       // buf p ready
        if (k0 + BK < K) { stageA(p ^ 1, k0 + BK); stageB(p ^ 1, k0 + BK); }

        // fragments per ISA 7.12.2: A row-striped per lane, B per-VGPR rows
        v16h a0, a1, b0, b1;
        const _Float16* ar0 = &As[p][(wm * 32 + lr) * LSTR];
        const _Float16* ar1 = &As[p][(wm * 32 + 16 + lr) * LSTR];
        #pragma unroll
        for (int i = 0; i < 8; ++i) { a0[i] = ar0[8 * lh + i];      a1[i] = ar1[8 * lh + i]; }
        #pragma unroll
        for (int i = 0; i < 8; ++i) { a0[8 + i] = ar0[16 + 8 * lh + i]; a1[8 + i] = ar1[16 + 8 * lh + i]; }
        const _Float16* br0 = &Bs[p][(wn * 32 + lr) * LSTR];
        const _Float16* br1 = &Bs[p][(wn * 32 + 16 + lr) * LSTR];
        #pragma unroll
        for (int i = 0; i < 16; ++i) { int kb = (i < 8) ? 8 * lh + i : 8 + 8 * lh + i; b0[i] = br0[kb]; b1[i] = br1[kb]; }

        acc00 = __builtin_amdgcn_wmma_f32_16x16x32_f16(false, a0, false, b0, (short)0, acc00, false, false);
        acc01 = __builtin_amdgcn_wmma_f32_16x16x32_f16(false, a0, false, b1, (short)0, acc01, false, false);
        acc10 = __builtin_amdgcn_wmma_f32_16x16x32_f16(false, a1, false, b0, (short)0, acc10, false, false);
        acc11 = __builtin_amdgcn_wmma_f32_16x16x32_f16(false, a1, false, b1, (short)0, acc11, false, false);
        p ^= 1;
    }

    // ---- epilogue ----
    const int colA = n0 + wn * 32 + lr;
    const int colB = colA + 16;
    #pragma unroll
    for (int f = 0; f < 2; ++f) {
        const v8f* aL = f ? &acc10 : &acc00;
        const v8f* aR = f ? &acc11 : &acc01;
        #pragma unroll
        for (int r = 0; r < 8; ++r) {
            int row = m0 + wm * 32 + f * 16 + lh * 8 + r;
            if (row >= M) continue;
            long long base = (long long)row * ldc;
            if (colA < N) {
                float v = alpha * (*aL)[r];
                if (bias)  v += bias[colA];
                if (resid) v += resid[base + colA];
                if (act)   v  = gelu_f(v);
                C[base + colA] = v;
            }
            if (colB < N) {
                float v = alpha * (*aR)[r];
                if (bias)  v += bias[colB];
                if (resid) v += resid[base + colB];
                if (act)   v  = gelu_f(v);
                C[base + colB] = v;
            }
        }
    }
}

// ---------------------------------------------------------------------------
// RevIN: per-batch mean/std over SEQ, write xn and stats
// ---------------------------------------------------------------------------
__global__ __launch_bounds__(256) void revin_kernel(
    const float* __restrict__ x, const float* __restrict__ rw, const float* __restrict__ rb,
    float* __restrict__ xn, float* __restrict__ stats)
{
    int b = blockIdx.x, tid = threadIdx.x;
    const float* xb = x + (long long)b * SEQL;
    float s = 0.0f, s2 = 0.0f;
    for (int i = tid; i < SEQL; i += 256) { float v = xb[i]; s += v; s2 += v * v; }
    __shared__ float r1[256], r2[256];
    r1[tid] = s; r2[tid] = s2; __syncthreads();
    for (int st = 128; st > 0; st >>= 1) {
        if (tid < st) { r1[tid] += r1[tid + st]; r2[tid] += r2[tid + st]; }
        __syncthreads();
    }
    __shared__ float mu, sd;
    if (tid == 0) {
        mu = r1[0] / (float)SEQL;
        float var = r2[0] / (float)SEQL - mu * mu;
        sd = sqrtf(var + 1e-5f);
        stats[b] = mu; stats[8 + b] = sd;
    }
    __syncthreads();
    float w = rw[0], bb = rb[0];
    for (int i = tid; i < SEQL; i += 256)
        xn[(long long)b * SEQL + i] = (xb[i] - mu) / sd * w + bb;
}

// ---------------------------------------------------------------------------
// Patch embedding -> d_out patches [B, D, N]
// ---------------------------------------------------------------------------
__global__ __launch_bounds__(256) void patch_embed_kernel(
    const float* __restrict__ xn, const float* __restrict__ pW,
    const float* __restrict__ pb, const float* __restrict__ pos,
    float* __restrict__ patchesT)
{
    long long idx = (long long)blockIdx.x * 256 + threadIdx.x;
    if (idx >= (long long)BSZ * DM * NP) return;
    int n = (int)(idx % NP);
    long long rest = idx / NP;
    int d = (int)(rest % DM);
    int b = (int)(rest / DM);
    const float* xb = xn + (long long)b * SEQL + (long long)n * 8;
    float acc = pb[d] + pos[(long long)n * DM + d];
    #pragma unroll
    for (int k = 0; k < 16; ++k) acc += xb[k] * pW[k * DM + d];
    patchesT[(long long)b * PT_SZ + (long long)d * NP + n] = acc;
}

// ---------------------------------------------------------------------------
// DWT (grouped stride-2 conv, pad 1) -> LL/LH (d_out) + LLseq/LHseq (ws)
// ---------------------------------------------------------------------------
__global__ __launch_bounds__(256) void dwt_kernel(
    const float* __restrict__ pT, const float* __restrict__ h4, const float* __restrict__ g4,
    float* __restrict__ LLd, float* __restrict__ LHd,
    float* __restrict__ LLseq, float* __restrict__ LHseq)
{
    long long idx = (long long)blockIdx.x * 256 + threadIdx.x;
    if (idx >= (long long)BSZ * LW * DM) return;
    int d = (int)(idx & (DM - 1));
    int t = (int)((idx >> 9) % LW);
    int b = (int)(idx / ((long long)LW * DM));
    const float* src = pT + (long long)b * PT_SZ + (long long)d * NP;
    float ah = 0.0f, ag = 0.0f;
    #pragma unroll
    for (int k = 0; k < 4; ++k) {
        int si = 2 * t - 1 + k;
        if (si >= 0 && si < NP) {
            float v = src[si];
            ah += v * h4[d * 4 + k];
            ag += v * g4[d * 4 + k];
        }
    }
    LLd[(long long)b * DM * LW + (long long)d * LW + t] = ah;
    LHd[(long long)b * DM * LW + (long long)d * LW + t] = ag;
    LLseq[(long long)b * LBD + (long long)t * DM + d] = ah;
    LHseq[(long long)b * LBD + (long long)t * DM + d] = ag;
}

// ---------------------------------------------------------------------------
// Energy gate: gate[b,t] = sigmoid((mean_d |LHseq| - sigmoid(raw_tau)) * 10)
// ---------------------------------------------------------------------------
__global__ __launch_bounds__(256) void gate_kernel(
    const float* __restrict__ LHseq, const float* __restrict__ raw_tau,
    float* __restrict__ gate)
{
    int row = blockIdx.x;
    int tid = threadIdx.x;
    const float* p = LHseq + (long long)row * DM;
    float s = fabsf(p[tid]) + fabsf(p[tid + 256]);
    __shared__ float red[256];
    red[tid] = s; __syncthreads();
    for (int st = 128; st > 0; st >>= 1) {
        if (tid < st) red[tid] += red[tid + st];
        __syncthreads();
    }
    if (tid == 0) {
        float tau = sigmoid_f(raw_tau[0]);
        gate[row] = sigmoid_f((red[0] / (float)DM - tau) * 10.0f);
    }
}

// ---------------------------------------------------------------------------
// Softmax over row of 511 (ld 512), optional column gate.  grid (LW, 64)
// ---------------------------------------------------------------------------
__global__ __launch_bounds__(256) void softmax_kernel(
    float* __restrict__ S, const float* __restrict__ gate)
{
    int l = blockIdx.x, z = blockIdx.y, tid = threadIdx.x;
    int bidx = z >> 3;
    float* row = S + (long long)z * LBD + (long long)l * 512;
    float v0 = (tid < LW)       ? row[tid]       : -3.0e38f;
    float v1 = (tid + 256 < LW) ? row[tid + 256] : -3.0e38f;
    __shared__ float red[256];
    red[tid] = fmaxf(v0, v1); __syncthreads();
    for (int st = 128; st > 0; st >>= 1) {
        if (tid < st) red[tid] = fmaxf(red[tid], red[tid + st]);
        __syncthreads();
    }
    float mx = red[0]; __syncthreads();
    float e0 = (tid < LW)       ? expf(v0 - mx) : 0.0f;
    float e1 = (tid + 256 < LW) ? expf(v1 - mx) : 0.0f;
    red[tid] = e0 + e1; __syncthreads();
    for (int st = 128; st > 0; st >>= 1) {
        if (tid < st) red[tid] += red[tid + st];
        __syncthreads();
    }
    float inv = 1.0f / red[0];
    if (tid < LW) {
        float w = e0 * inv;
        if (gate) w *= gate[(long long)bidx * LW + tid];
        row[tid] = w;
    }
    if (tid + 256 < LW) {
        float w = e1 * inv;
        if (gate) w *= gate[(long long)bidx * LW + tid + 256];
        row[tid + 256] = w;
    }
    if (tid == 0) row[LW] = 0.0f;
}

// ---------------------------------------------------------------------------
// LayerNorm of x1 + x2 (+ x3) over D=512; grid = MROWS
// ---------------------------------------------------------------------------
__global__ __launch_bounds__(256) void ln3_kernel(
    const float* __restrict__ x1, const float* __restrict__ x2, const float* __restrict__ x3,
    const float* __restrict__ g, const float* __restrict__ bt, float* __restrict__ out)
{
    long long base = (long long)blockIdx.x * DM;
    int tid = threadIdx.x;
    float v0 = x1[base + tid]       + x2[base + tid]       + (x3 ? x3[base + tid]       : 0.0f);
    float v1 = x1[base + tid + 256] + x2[base + tid + 256] + (x3 ? x3[base + tid + 256] : 0.0f);
    __shared__ float r1[256], r2[256];
    r1[tid] = v0 + v1; r2[tid] = v0 * v0 + v1 * v1; __syncthreads();
    for (int st = 128; st > 0; st >>= 1) {
        if (tid < st) { r1[tid] += r1[tid + st]; r2[tid] += r2[tid + st]; }
        __syncthreads();
    }
    float mu = r1[0] / (float)DM;
    float var = r2[0] / (float)DM - mu * mu;
    float rstd = rsqrtf(var + 1e-5f);
    out[base + tid]       = (v0 - mu) * rstd * g[tid]       + bt[tid];
    out[base + tid + 256] = (v1 - mu) * rstd * g[tid + 256] + bt[tid + 256];
}

// ---------------------------------------------------------------------------
// Head: gated fusion + [8,523264]@[523264,96] + RevIN denorm -> preds
// ---------------------------------------------------------------------------
__global__ __launch_bounds__(256) void head_kernel(
    const float* __restrict__ LLo, const float* __restrict__ LHo,
    const float* __restrict__ llg, const float* __restrict__ lhg,
    const float* __restrict__ hW, const float* __restrict__ hb,
    const float* __restrict__ rw, const float* __restrict__ rb,
    const float* __restrict__ stats, float* __restrict__ preds)
{
    int n = blockIdx.x, tid = threadIdx.x;
    float acc[8];
    #pragma unroll
    for (int b2 = 0; b2 < 8; ++b2) acc[b2] = 0.0f;
    const long long KH = LBD;
    for (long long j = tid; j < KH; j += 256) {
        float w = hW[j * PREDN + n] * llg[j];
        #pragma unroll
        for (int b2 = 0; b2 < 8; ++b2) acc[b2] += LLo[(long long)b2 * KH + j] * w;
    }
    for (long long j = tid; j < KH; j += 256) {
        float w = hW[(KH + j) * PREDN + n] * lhg[j];
        #pragma unroll
        for (int b2 = 0; b2 < 8; ++b2) acc[b2] += LHo[(long long)b2 * KH + j] * w;
    }
    __shared__ float red[256];
    for (int b2 = 0; b2 < 8; ++b2) {
        red[tid] = acc[b2]; __syncthreads();
        for (int st = 128; st > 0; st >>= 1) {
            if (tid < st) red[tid] += red[tid + st];
            __syncthreads();
        }
        if (tid == 0) {
            float raw = red[0] + hb[n];
            preds[b2 * PREDN + n] =
                (raw - rb[0]) / (rw[0] + 1e-10f) * stats[8 + b2] + stats[b2];
        }
        __syncthreads();
    }
}

// ---------------------------------------------------------------------------
// Inverse DWT (transposed grouped conv) -> patch_recon [B, D, 1022]
// ---------------------------------------------------------------------------
__global__ __launch_bounds__(256) void idwt_kernel(
    const float* __restrict__ LLo, const float* __restrict__ LHo,
    const float* __restrict__ h4, const float* __restrict__ g4,
    float* __restrict__ recon)
{
    long long idx = (long long)blockIdx.x * 256 + threadIdx.x;
    const long long TOT = (long long)BSZ * DM * 1022;
    if (idx >= TOT) return;
    int i = (int)(idx % 1022);
    long long rest = idx / 1022;
    int d = (int)(rest % DM);
    int b = (int)(rest / DM);
    float acc = 0.0f;
    #pragma unroll
    for (int k = 0; k < 4; ++k) {
        int m = i + k - 2;
        if (m >= 0 && m <= 1020 && (m & 1) == 0) {
            int t = m >> 1;
            long long off = (long long)b * LBD + (long long)t * DM + d;
            acc += h4[d * 4 + (3 - k)] * LLo[off];
            acc += g4[d * 4 + (3 - k)] * LHo[off];
        }
    }
    recon[(long long)b * DM * 1022 + (long long)d * 1022 + i] = acc;
}

// ---------------------------------------------------------------------------
// Host-side launcher
// ---------------------------------------------------------------------------
static void launch_gemm(hipStream_t s,
    const float* A, long long lda, long long sAo, long long sAi,
    const float* B, long long ldb, long long sBo, long long sBi, int transB,
    const float* bias, const float* resid,
    float* C, long long ldc, long long sCo, long long sCi,
    int M, int N, int K, float alpha, int act, int HHd, int Z)
{
    dim3 grid((N + BN - 1) / BN, (M + BM - 1) / BM, Z);
    gemm_wmma_kernel<<<grid, 256, 0, s>>>(A, lda, sAo, sAi, B, ldb, sBo, sBi, transB,
                                          bias, resid, C, ldc, sCo, sCi,
                                          M, N, K, alpha, act, HHd);
}

extern "C" void kernel_launch(void* const* d_in, const int* in_sizes, int n_in,
                              void* d_out, int out_size, void* d_ws, size_t ws_size,
                              hipStream_t stream) {
    (void)in_sizes; (void)n_in; (void)out_size; (void)ws_size;
    const float* x       = (const float*)d_in[0];
    const float* revin_w = (const float*)d_in[1];
    const float* revin_b = (const float*)d_in[2];
    const float* patch_W = (const float*)d_in[3];
    const float* patch_b = (const float*)d_in[4];
    const float* pos_emb = (const float*)d_in[5];
    const float* dwt_h   = (const float*)d_in[6];
    const float* dwt_g   = (const float*)d_in[7];
    const float* raw_tau = (const float*)d_in[8];
    const float* attn_W  = (const float*)d_in[9];
    const float* attn_b  = (const float*)d_in[10];
    const float* mlp_W1  = (const float*)d_in[11];
    const float* mlp_b1  = (const float*)d_in[12];
    const float* mlp_W2  = (const float*)d_in[13];
    const float* mlp_b2  = (const float*)d_in[14];
    const float* ln_g    = (const float*)d_in[15];
    const float* ln_b    = (const float*)d_in[16];
    const float* ll_gate = (const float*)d_in[17];
    const float* lh_gate = (const float*)d_in[18];
    const float* head_W  = (const float*)d_in[19];
    const float* head_b  = (const float*)d_in[20];

    float* out = (float*)d_out;
    float* ws  = (float*)d_ws;

    // 1) RevIN
    revin_kernel<<<BSZ, 256, 0, stream>>>(x, revin_w, revin_b, ws + W_XN, ws + W_STATS);

    // 2) Patch embedding -> patches [B,D,N]
    {
        long long tot = (long long)BSZ * DM * NP;
        patch_embed_kernel<<<(unsigned)((tot + 255) / 256), 256, 0, stream>>>(
            ws + W_XN, patch_W, patch_b, pos_emb, out + O_PATCH);
    }

    // 3) DWT
    {
        long long tot = (long long)BSZ * LW * DM;
        dwt_kernel<<<(unsigned)((tot + 255) / 256), 256, 0, stream>>>(
            out + O_PATCH, dwt_h, dwt_g, out + O_LL, out + O_LH,
            ws + W_LLSEQ, ws + W_LHSEQ);
    }

    // 4) Energy gate
    gate_kernel<<<(unsigned)MROWS, 256, 0, stream>>>(ws + W_LHSEQ, raw_tau, ws + W_GATE);

    // 5) Three attentions
    const long long DD = (long long)DM * DM;
    const float* qx[3]    = { ws + W_LLSEQ, ws + W_LHSEQ, ws + W_LLOUT };
    const float* kvx[3]   = { ws + W_LLSEQ, ws + W_LHSEQ, ws + W_LHOUT };
    float*       dest[3]  = { ws + W_LLOUT, ws + W_LHOUT, ws + W_CROSS };
    const float* gates[3] = { nullptr, ws + W_GATE, nullptr };

    for (int ai = 0; ai < 3; ++ai) {
        const float* W  = attn_W + (long long)ai * 4 * DD;
        const float* bb = attn_b + (long long)ai * 4 * DM;
        launch_gemm(stream, qx[ai], DM, 0, 0, W + 0 * DD, DM, 0, 0, 0,
                    bb + 0 * DM, nullptr, ws + W_Q, DM, 0, 0,
                    (int)MROWS, DM, DM, 1.0f, 0, 1, 1);
        launch_gemm(stream, kvx[ai], DM, 0, 0, W + 1 * DD, DM, 0, 0, 0,
                    bb + 1 * DM, nullptr, ws + W_K, DM, 0, 0,
                    (int)MROWS, DM, DM, 1.0f, 0, 1, 1);
        launch_gemm(stream, kvx[ai], DM, 0, 0, W + 2 * DD, DM, 0, 0, 0,
                    bb + 2 * DM, nullptr, ws + W_V, DM, 0, 0,
                    (int)MROWS, DM, DM, 1.0f, 0, 1, 1);
        // scores = Q K^T / 8  (z = b*8 + h, head slice = 64 cols)
        launch_gemm(stream, ws + W_Q, DM, LBD, 64,
                    ws + W_K, DM, LBD, 64, 1,
                    nullptr, nullptr,
                    ws + W_S, 512, 8 * LBD, LBD,
                    LW, LW, 64, 0.125f, 0, HH_, BSZ * HH_);
        {
            dim3 gs(LW, BSZ * HH_);
            softmax_kernel<<<gs, 256, 0, stream>>>(ws + W_S, gates[ai]);
        }
        // O = softmax(S) V  -> merged-head layout
        launch_gemm(stream, ws + W_S, 512, 8 * LBD, LBD,
                    ws + W_V, DM, LBD, 64, 0,
                    nullptr, nullptr,
                    ws + W_O, DM, LBD, 64,
                    LW, 64, LW, 1.0f, 0, HH_, BSZ * HH_);
        launch_gemm(stream, ws + W_O, DM, 0, 0, W + 3 * DD, DM, 0, 0, 0,
                    bb + 3 * DM, nullptr, dest[ai], DM, 0, 0,
                    (int)MROWS, DM, DM, 1.0f, 0, 1, 1);
    }

    // 6) Fused layernorms
    ln3_kernel<<<(unsigned)MROWS, 256, 0, stream>>>(
        ws + W_LLSEQ, ws + W_LLOUT, ws + W_CROSS, ln_g, ln_b, ws + W_F1);
    ln3_kernel<<<(unsigned)MROWS, 256, 0, stream>>>(
        ws + W_LHSEQ, ws + W_LHOUT, nullptr, ln_g + DM, ln_b + DM, ws + W_F2);

    // 7) MLPs (bias+GELU fused; second GEMM fuses residual)
    launch_gemm(stream, ws + W_F1, DM, 0, 0, mlp_W1, DFF_, 0, 0, 0,
                mlp_b1, nullptr, ws + W_HID, DFF_, 0, 0,
                (int)MROWS, DFF_, DM, 1.0f, 1, 1, 1);
    launch_gemm(stream, ws + W_HID, DFF_, 0, 0, mlp_W2, DM, 0, 0, 0,
                mlp_b2, ws + W_F1, ws + W_LLO, DM, 0, 0,
                (int)MROWS, DM, DFF_, 1.0f, 0, 1, 1);
    launch_gemm(stream, ws + W_F2, DM, 0, 0, mlp_W1 + (long long)DM * DFF_, DFF_, 0, 0, 0,
                mlp_b1 + DFF_, nullptr, ws + W_HID, DFF_, 0, 0,
                (int)MROWS, DFF_, DM, 1.0f, 1, 1, 1);
    launch_gemm(stream, ws + W_HID, DFF_, 0, 0, mlp_W2 + (long long)DFF_ * DM, DM, 0, 0, 0,
                mlp_b2 + DM, ws + W_F2, ws + W_LHO, DM, 0, 0,
                (int)MROWS, DM, DFF_, 1.0f, 0, 1, 1);

    // 8) Head -> preds
    head_kernel<<<PREDN, 256, 0, stream>>>(
        ws + W_LLO, ws + W_LHO, ll_gate, lh_gate, head_W, head_b,
        revin_w, revin_b, ws + W_STATS, out + O_PRED);

    // 9) Inverse DWT -> patch_recon
    {
        long long tot = (long long)BSZ * DM * 1022;
        idwt_kernel<<<(unsigned)((tot + 255) / 256), 256, 0, stream>>>(
            ws + W_LLO, ws + W_LHO, dwt_h, dwt_g, out + O_RECON);
    }
}
